// MultiheadSelfAttention_63273458204728
// MI455X (gfx1250) — matrix operand order repaired and checked by reading, not compiled, not run on hardware
//
#include <hip/hip_runtime.h>

#define D_MODEL 1024
#define NUM_HEADS 16
#define D_K 64
#define SEQ 2048
#define BATCH 2
#define M_TOKENS (BATCH * SEQ) /* 4096 */
#define KSTEP 64               /* K tile staged in LDS per buffer */

typedef __attribute__((ext_vector_type(16))) _Float16 v16h;
typedef __attribute__((ext_vector_type(8)))  _Float16 v8h;
typedef __attribute__((ext_vector_type(8)))  float    v8f;

// A-operand fragment (16x32 f16, MxK): lane L holds row (L&15), K-chunks
// {kg*8..+7} and {16+kg*8..+7}, kg = L>>4.  p -> first chunk; second at +16.
static __device__ __forceinline__ v16h lda_frag(const _Float16* p) {
  v8h lo = *(const v8h*)(p);
  v8h hi = *(const v8h*)(p + 16);
  return __builtin_shufflevector(lo, hi, 0, 1, 2, 3, 4, 5, 6, 7,
                                 8, 9, 10, 11, 12, 13, 14, 15);
}

// B-operand fragment (32x16 f16, KxN): lane L holds B^T row (L&15),
// K-chunks {kg*8..+7} and {16+kg*8..+7}.  p -> first chunk; second at +16.
static __device__ __forceinline__ v16h ldb_frag(const _Float16* p) {
  v8h lo = *(const v8h*)(p);
  v8h hi = *(const v8h*)(p + 16);
  return __builtin_shufflevector(lo, hi, 0, 1, 2, 3, 4, 5, 6, 7,
                                 8, 9, 10, 11, 12, 13, 14, 15);
}

static __device__ __forceinline__ v8f wmma_f16(v16h a, v16h b, v8f c) {
  return __builtin_amdgcn_wmma_f32_16x16x32_f16(false, a, false, b,
                                                (short)0, c, false, false);
}

// Async DMA: 16 bytes per lane, global -> LDS, tracked by ASYNCcnt.
static __device__ __forceinline__ void async_ld_b128(const void* gaddr,
                                                     const _Float16* lds_ptr) {
  const unsigned lds_off = (unsigned)(unsigned long long)lds_ptr;  // [31:0] = LDS offset
  asm volatile("global_load_async_to_lds_b128 %0, %1, off"
               :
               : "v"(lds_off), "v"(gaddr)
               : "memory");
}

__global__ void cvt_f32_to_f16(const float* __restrict__ in,
                               _Float16* __restrict__ out, int n) {
  int i = blockIdx.x * blockDim.x + threadIdx.x;
  if (i < n) out[i] = (_Float16)in[i];
}

// Y = X * W^T  (einsum 'bsi,oi->bso'), M=4096, N=K=1024.
// Block = 4 waves, output tile 128(M) x 64(N); each wave owns 32x64.
// B tile (64 rows x 64 K-halfs) double-buffered in LDS via async DMA.
// mode 0: Q-proj + RoPE -> outh[b,h,s,d]   (f16)
// mode 1: K-proj + RoPE -> outh[b,h,s,d]   (f16)
// mode 2: V-proj        -> outh[b,h,d,s]   (f16, transposed for P*V)
// mode 3: O-proj        -> outf[row, col]  (f32, final result)
__global__ __launch_bounds__(128) void proj_gemm(
    const _Float16* __restrict__ X, const _Float16* __restrict__ W,
    const int* __restrict__ pos, _Float16* __restrict__ outh,
    float* __restrict__ outf, int mode) {
  __shared__ __align__(16) _Float16 Bsh[2][64 * KSTEP];  // 2 x 8 KB

  const int lane = threadIdx.x & 31;
  const int wave = threadIdx.x >> 5;
  const int lr = lane & 15;
  const int kg = lane >> 4;
  const int row0 = blockIdx.y * 128 + wave * 32;
  const int col0 = blockIdx.x * 64;

  // Stage 64 (N rows) x 128B (64 halfs of K) of W into Bsh[buf].
  // 8 KB = 512 x 16B chunks = 128 threads x 4 async ops (4 per wave).
  auto stage_B = [&](int buf, int k0) {
#pragma unroll
    for (int r = 0; r < 4; ++r) {
      const int idx = r * 128 + threadIdx.x;  // 0..511
      const int row = idx >> 3;               // 0..63
      const int chunk = idx & 7;              // 16B chunk within row
      async_ld_b128(W + (size_t)(col0 + row) * D_MODEL + k0 + chunk * 8,
                    &Bsh[buf][row * KSTEP + chunk * 8]);
    }
  };

  v8f acc[8] = {};
  const _Float16* Arow0 = X + (size_t)(row0 + lr) * D_MODEL;
  const _Float16* Arow1 = X + (size_t)(row0 + 16 + lr) * D_MODEL;

  stage_B(0, 0);
  for (int ks = 0; ks < D_MODEL / KSTEP; ++ks) {
    const int k0 = ks * KSTEP;
    if (ks + 1 < D_MODEL / KSTEP) {
      stage_B((ks + 1) & 1, k0 + KSTEP);
      // in-order completion: <=4 outstanding => current buffer's 4 are done
      asm volatile("s_wait_asynccnt 4" ::: "memory");
    } else {
      asm volatile("s_wait_asynccnt 0" ::: "memory");
    }
    __syncthreads();  // all waves' DMA into current buffer visible

    const _Float16* Bbuf = &Bsh[ks & 1][0];
#pragma unroll
    for (int kk = 0; kk < 2; ++kk) {  // two 32-K substeps per buffer
      v16h a0 = lda_frag(Arow0 + k0 + kk * 32 + kg * 8);
      v16h a1 = lda_frag(Arow1 + k0 + kk * 32 + kg * 8);
#pragma unroll
      for (int t = 0; t < 4; ++t) {
        v16h bf = ldb_frag(Bbuf + (t * 16 + lr) * KSTEP + kk * 32 + kg * 8);
        acc[t] = wmma_f16(a0, bf, acc[t]);
        acc[4 + t] = wmma_f16(a1, bf, acc[4 + t]);
      }
    }
    __syncthreads();  // reads done before this buffer is refilled
  }

#pragma unroll
  for (int s2 = 0; s2 < 2; ++s2) {
#pragma unroll
    for (int t = 0; t < 4; ++t) {
      const int col = col0 + t * 16 + lr;
      float vals[8];
#pragma unroll
      for (int i = 0; i < 8; ++i) vals[i] = acc[s2 * 4 + t][i];

      if (mode <= 1) {
        // RoPE: (even,odd) dim pairs are adjacent columns -> adjacent lanes.
        const int kidx = (col & (D_K - 1)) >> 1;
        // inv_freq = 10000^(-2k/64) = exp(-k * ln(10000)/32)
        const float inv_freq = __expf(-0.2878231366242557f * (float)kidx);
        const bool odd = (col & 1) != 0;
#pragma unroll
        for (int i = 0; i < 8; ++i) {
          const int row = row0 + s2 * 16 + i + kg * 8;
          const float p = (float)pos[row & (SEQ - 1)];
          float sn, cs;
          __sincosf(p * inv_freq, &sn, &cs);
          const float partner = __shfl_xor(vals[i], 1, 32);
          vals[i] = odd ? (partner * sn + vals[i] * cs)
                        : (vals[i] * cs - partner * sn);
        }
      }

      const int h = col >> 6;
      const int d = col & (D_K - 1);
#pragma unroll
      for (int i = 0; i < 8; ++i) {
        const int row = row0 + s2 * 16 + i + kg * 8;
        const int bb = row >> 11;        // row / SEQ
        const int ss = row & (SEQ - 1);  // row % SEQ
        if (mode == 3) {
          outf[(size_t)row * D_MODEL + col] = vals[i];
        } else if (mode == 2) {
          outh[((size_t)(bb * NUM_HEADS + h) * D_K + d) * SEQ + ss] =
              (_Float16)vals[i];
        } else {
          outh[((size_t)(bb * NUM_HEADS + h) * SEQ + ss) * D_K + d] =
              (_Float16)vals[i];
        }
      }
    }
  }
}

// Causal flash attention. One wave owns 16 query rows; block = 4 waves,
// grid = (SEQ/64, H, B).  No block-wide barriers (waves have different
// causal trip counts); LDS is partitioned per wave.
__global__ __launch_bounds__(128) void flash_attn(
    const _Float16* __restrict__ xq, const _Float16* __restrict__ xk,
    const _Float16* __restrict__ xvt, _Float16* __restrict__ attnh) {
  __shared__ __align__(32) _Float16 ldsP[4][16 * 32];
  const int lane = threadIdx.x & 31;
  const int wave = threadIdx.x >> 5;
  const int lr = lane & 15;
  const int kg = lane >> 4;
  const int b = blockIdx.z, h = blockIdx.y;
  const int bh = b * NUM_HEADS + h;
  const int qbase = blockIdx.x * 64 + wave * 16;

  const _Float16* Qb = xq + (size_t)bh * SEQ * D_K;
  const _Float16* Kb = xk + (size_t)bh * SEQ * D_K;
  const _Float16* Vt = xvt + (size_t)bh * D_K * SEQ;

  v16h qf[2];
  {
    const _Float16* qrow = Qb + (size_t)(qbase + lr) * D_K;
    qf[0] = lda_frag(qrow + kg * 8);
    qf[1] = lda_frag(qrow + 32 + kg * 8);
  }

  float m[8], l[8];
  v8f acc[4] = {};
#pragma unroll
  for (int i = 0; i < 8; ++i) { m[i] = -1e30f; l[i] = 0.f; }

  const int kend = qbase + 16;  // max key needed (exclusive) for this wave
  for (int k0 = 0; k0 < kend; k0 += 32) {
    v8f s0 = {}, s1 = {};
#pragma unroll
    for (int kk = 0; kk < 2; ++kk) {
      const int kt = k0 + kk * 16;
      const _Float16* krow = Kb + (size_t)(kt + lr) * D_K;
      v8f sc = {};
      sc = wmma_f16(qf[0], ldb_frag(krow + kg * 8), sc);
      sc = wmma_f16(qf[1], ldb_frag(krow + 32 + kg * 8), sc);
      const int key = kt + lr;
#pragma unroll
      for (int i = 0; i < 8; ++i) {
        const int qi = qbase + i + kg * 8;
        const float v = sc[i] * 0.125f;  // 1/sqrt(D_K)
        sc[i] = (key <= qi) ? v : -1e30f;
      }
      if (kk == 0) s0 = sc; else s1 = sc;
    }

    // online softmax (rows per lane: row = i + 8*kg; cols across 16 lanes)
    float mloc[8], rs[8], alpha[8];
    v8f p0 = {}, p1 = {};
#pragma unroll
    for (int i = 0; i < 8; ++i) mloc[i] = fmaxf(s0[i], s1[i]);
#pragma unroll
    for (int off = 1; off < 16; off <<= 1)
#pragma unroll
      for (int i = 0; i < 8; ++i)
        mloc[i] = fmaxf(mloc[i], __shfl_xor(mloc[i], off, 32));
#pragma unroll
    for (int i = 0; i < 8; ++i) {
      const float mn = fmaxf(m[i], mloc[i]);
      alpha[i] = __expf(m[i] - mn);
      m[i] = mn;
      p0[i] = __expf(s0[i] - mn);
      p1[i] = __expf(s1[i] - mn);
      rs[i] = p0[i] + p1[i];
    }
#pragma unroll
    for (int off = 1; off < 16; off <<= 1)
#pragma unroll
      for (int i = 0; i < 8; ++i) rs[i] += __shfl_xor(rs[i], off, 32);
#pragma unroll
    for (int i = 0; i < 8; ++i) l[i] = l[i] * alpha[i] + rs[i];
#pragma unroll
    for (int t = 0; t < 4; ++t)
#pragma unroll
      for (int i = 0; i < 8; ++i) acc[t][i] *= alpha[i];

    // C-layout P -> LDS (row-major 16x32) -> reload as A-fragment
    _Float16* P = &ldsP[wave][0];
#pragma unroll
    for (int i = 0; i < 8; ++i) {
      const int r = i + kg * 8;
      P[r * 32 + lr] = (_Float16)p0[i];
      P[r * 32 + 16 + lr] = (_Float16)p1[i];
    }
    v16h pf = lda_frag(P + lr * 32 + kg * 8);
#pragma unroll
    for (int t = 0; t < 4; ++t) {
      const _Float16* vrow = Vt + (size_t)(t * 16 + lr) * SEQ + k0;
      acc[t] = wmma_f16(pf, ldb_frag(vrow + kg * 8), acc[t]);
    }
  }

  // normalize and store as f16 rows [b, s, h*64+d] for the output projection
#pragma unroll
  for (int i = 0; i < 8; ++i) {
    const int qi = qbase + i + kg * 8;
    const float rinv = 1.0f / l[i];
#pragma unroll
    for (int t = 0; t < 4; ++t) {
      attnh[((size_t)b * SEQ + qi) * D_MODEL + h * D_K + t * 16 + lr] =
          (_Float16)(acc[t][i] * rinv);
    }
  }
}

extern "C" void kernel_launch(void* const* d_in, const int* in_sizes, int n_in,
                              void* d_out, int out_size, void* d_ws,
                              size_t ws_size, hipStream_t stream) {
  (void)in_sizes; (void)n_in; (void)out_size; (void)ws_size;
  const float* Q = (const float*)d_in[0];
  const float* K = (const float*)d_in[1];
  const float* V = (const float*)d_in[2];
  const int* pos = (const int*)d_in[3];
  const float* wq = (const float*)d_in[4];
  const float* wk = (const float*)d_in[5];
  const float* wv = (const float*)d_in[6];
  const float* wo = (const float*)d_in[7];
  float* out = (float*)d_out;

  const size_t NT = (size_t)M_TOKENS * D_MODEL;  // 4 Mi elements
  const size_t NW = (size_t)D_MODEL * D_MODEL;   // 1 Mi elements
  _Float16* w16 = (_Float16*)d_ws;
  _Float16* Qh = w16;
  _Float16* Kh = Qh + NT;
  _Float16* Vh = Kh + NT;
  _Float16* attnh = Vh + NT;
  _Float16* xq = attnh + NT;
  _Float16* xk = xq + NT;
  _Float16* xvt = xk + NT;
  _Float16* wqh = xvt + NT;
  _Float16* wkh = wqh + NW;
  _Float16* wvh = wkh + NW;
  _Float16* woh = wvh + NW;  // total: 7*NT + 4*NW halfs = 64 MB

  const int cvtB = 256;
  cvt_f32_to_f16<<<dim3((unsigned)((NT + cvtB - 1) / cvtB)), dim3(cvtB), 0, stream>>>(Q, Qh, (int)NT);
  cvt_f32_to_f16<<<dim3((unsigned)((NT + cvtB - 1) / cvtB)), dim3(cvtB), 0, stream>>>(K, Kh, (int)NT);
  cvt_f32_to_f16<<<dim3((unsigned)((NT + cvtB - 1) / cvtB)), dim3(cvtB), 0, stream>>>(V, Vh, (int)NT);
  cvt_f32_to_f16<<<dim3((unsigned)((NW + cvtB - 1) / cvtB)), dim3(cvtB), 0, stream>>>(wq, wqh, (int)NW);
  cvt_f32_to_f16<<<dim3((unsigned)((NW + cvtB - 1) / cvtB)), dim3(cvtB), 0, stream>>>(wk, wkh, (int)NW);
  cvt_f32_to_f16<<<dim3((unsigned)((NW + cvtB - 1) / cvtB)), dim3(cvtB), 0, stream>>>(wv, wvh, (int)NW);
  cvt_f32_to_f16<<<dim3((unsigned)((NW + cvtB - 1) / cvtB)), dim3(cvtB), 0, stream>>>(wo, woh, (int)NW);

  dim3 pgrid(D_MODEL / 64, M_TOKENS / 128);  // (16, 32)
  dim3 pblk(128);
  proj_gemm<<<pgrid, pblk, 0, stream>>>(Qh, wqh, pos, xq, nullptr, 0);
  proj_gemm<<<pgrid, pblk, 0, stream>>>(Kh, wkh, pos, xk, nullptr, 1);
  proj_gemm<<<pgrid, pblk, 0, stream>>>(Vh, wvh, pos, xvt, nullptr, 2);

  flash_attn<<<dim3(SEQ / 64, NUM_HEADS, BATCH), pblk, 0, stream>>>(xq, xk, xvt, attnh);

  proj_gemm<<<pgrid, pblk, 0, stream>>>(attnh, woh, pos, nullptr, out, 3);
}
